// CrystalGraphConvNet_29918742184070
// MI455X (gfx1250) — matrix operand order repaired and checked by reading, not compiled, not run on hardware
//
#include <hip/hip_runtime.h>
#include <hip/hip_bf16.h>

// ---------------- problem constants ----------------
constexpr int   N_    = 100000;
constexpr int   M_    = 12;
constexpr int   N0_   = 2500;
constexpr int   ORIG_ = 92;
constexpr int   C_    = 64;        // hidden
constexpr int   NBR_  = 41;
constexpr int   H_    = 128;
constexpr int   NCONV_= 3;
constexpr int   KIN_  = 2*C_ + NBR_;     // 169
constexpr int   KPAD_ = 192;             // 6 k-steps of 32
constexpr int   KT_   = KPAD_/32;        // 6
constexpr int   CT_   = (2*C_)/16;       // 8 col tiles
constexpr long  NM_   = (long)N_*M_;     // 1,200,000
constexpr int   NTILE_= (int)(NM_/16);   // 75,000
constexpr int   NSLICE= 256;
constexpr float EPSV  = 1e-5f;

// ---------------- workspace layout (bytes) ----------------
constexpr size_t OFF_G   = 0;
constexpr size_t SZ_G    = (size_t)NM_*2*C_*sizeof(float);          // 614.4 MB
constexpr size_t OFF_SUM = OFF_G + SZ_G;
constexpr size_t SZ_SUM  = (size_t)N_*C_*sizeof(float);
constexpr size_t OFF_H   = OFF_SUM + SZ_SUM;
constexpr size_t SZ_H    = (size_t)N_*C_*sizeof(float);
constexpr size_t OFF_H16 = OFF_H + SZ_H;
constexpr size_t SZ_H16  = (size_t)N_*C_*2;
constexpr size_t OFF_WP  = OFF_H16 + SZ_H16;                        // 16B aligned
constexpr size_t SZ_WP   = (size_t)NCONV_*KT_*CT_*32*16*2;          // f16 fragments
constexpr size_t OFF_P1  = OFF_WP + SZ_WP;
constexpr size_t SZ_P1   = (size_t)NSLICE*2*(2*C_)*sizeof(float);
constexpr size_t OFF_P2  = OFF_P1 + SZ_P1;
constexpr size_t SZ_P2   = (size_t)NSLICE*2*C_*sizeof(float);
constexpr size_t OFF_SC1 = OFF_P2 + SZ_P2;
constexpr size_t SZ_SC1  = (size_t)2*(2*C_)*sizeof(float);
constexpr size_t OFF_SC2 = OFF_SC1 + SZ_SC1;
constexpr size_t SZ_SC2  = (size_t)2*C_*sizeof(float);
constexpr size_t OFF_CRY = OFF_SC2 + SZ_SC2;
constexpr size_t SZ_CRY  = (size_t)N0_*C_*sizeof(float);
constexpr size_t OFF_CNT = OFF_CRY + SZ_CRY;
constexpr size_t SZ_CNT  = (size_t)N0_*sizeof(float);
constexpr size_t OFF_Z   = OFF_CNT + SZ_CNT;

typedef __attribute__((ext_vector_type(16))) _Float16 v16h;
typedef __attribute__((ext_vector_type(8)))  _Float16 v8h;
typedef __attribute__((ext_vector_type(8)))  float    v8f;

__device__ __forceinline__ float softplusf(float x) {
    return fmaxf(x, 0.f) + log1pf(expf(-fabsf(x)));
}
__device__ __forceinline__ float sigmoidf_(float x) {
    return 1.f / (1.f + expf(-x));
}

// ---------------- pack fc_W into WMMA B-fragment layout (f16) ----------------
// index: (((l*KT + kt)*CT + ct)*32 + lane)*16 + j
// B 32x16 f16 layout: col = lane&15 ; k = kt*32 + 8*(lane>>4) + 16*(j>>3) + (j&7)
__global__ void packW_kernel(const float* __restrict__ fcW, _Float16* __restrict__ wp) {
    int id = blockIdx.x * 256 + threadIdx.x;              // 3*6*8*32 = 4608
    if (id >= NCONV_*KT_*CT_*32) return;
    int lane = id & 31;
    int q    = id >> 5;
    int ct   = q & 7;
    int q2   = q >> 3;           // l*KT + kt
    int kt   = q2 % KT_;
    int l    = q2 / KT_;
    int col  = ct*16 + (lane & 15);
    int kb   = kt*32 + 8*(lane >> 4);
    _Float16* dst = wp + (size_t)id * 16;
#pragma unroll
    for (int j = 0; j < 16; ++j) {
        int k = kb + 16*(j >> 3) + (j & 7);
        float v = (k < KIN_) ? fcW[((size_t)l*KIN_ + k)*(2*C_) + col] : 0.f;
        dst[j] = (_Float16)v;
    }
}

// ---------------- embedding: h = atom_fea @ emb_W + emb_b ----------------
__global__ void embed_kernel(const float* __restrict__ af, const float* __restrict__ W,
                             const float* __restrict__ b, float* __restrict__ h,
                             _Float16* __restrict__ h16) {
    int idx = blockIdx.x * 256 + threadIdx.x;             // N*C exactly
    int atom = idx >> 6, c = idx & 63;
    const float* a = af + (size_t)atom * ORIG_;
    float acc = b[c];
#pragma unroll 4
    for (int k = 0; k < ORIG_; ++k) acc += a[k] * W[(size_t)k*C_ + c];
    h[idx]   = acc;
    h16[idx] = (_Float16)acc;
}

// ---------------- conv fc GEMM: g = [h_self | h_nbr | nbr_fea] @ W + b ------
// block = 256 threads = 8 waves; one 16-row tile; wave w -> col-tile w.
__global__ void __launch_bounds__(256)
conv_fc_kernel(const _Float16* __restrict__ h16, const int* __restrict__ nidx,
               const float* __restrict__ nfea, const _Float16* __restrict__ wp,
               const float* __restrict__ fcb, float* __restrict__ g,
               float* __restrict__ p1) {
    __shared__ __align__(16) _Float16 At[16][200];   // 200-half stride: bank-safe

    const int tile = blockIdx.x;
    const int t    = threadIdx.x;
    // ---- stage gathered A tile into LDS ----
    {
        int row  = t >> 4;
        int kl   = t & 15;
        long grow = (long)tile*16 + row;
        int atom = (int)(grow / M_);
        int m    = (int)(grow % M_);
        int nbr  = nidx[atom*M_ + m];
        const float* nf = nfea + (size_t)(atom*M_ + m) * NBR_;
#pragma unroll
        for (int i = 0; i < KPAD_/16; ++i) {
            int k = kl + 16*i;
            _Float16 v;
            if      (k < C_)    v = h16[(size_t)atom*C_ + k];
            else if (k < 2*C_)  v = h16[(size_t)nbr*C_ + (k - C_)];
            else if (k < KIN_)  v = (_Float16)nf[k - 2*C_];
            else                v = (_Float16)0.f;
            At[row][k] = v;
        }
    }
    __syncthreads();

    const int wave = t >> 5;          // col tile 0..7
    const int lane = t & 31;
    const int arow = lane & 15;
    const int kup  = (lane >> 4) * 8;
    const int col  = wave*16 + (lane & 15);

    v8f acc = {};
#pragma unroll
    for (int kt = 0; kt < KT_; ++kt) {
        // A fragment: k = kt*32 + 16*(j>>3) + kup + (j&7)
        v8h alo = *(const v8h*)&At[arow][kt*32 + kup];
        v8h ahi = *(const v8h*)&At[arow][kt*32 + kup + 16];
        v16h a, b;
#pragma unroll
        for (int j = 0; j < 8; ++j) { a[j] = alo[j]; a[j+8] = ahi[j]; }
        const _Float16* bp = wp + ((size_t)(kt*CT_ + wave)*32 + lane)*16;
        v8h blo = *(const v8h*)bp;
        v8h bhi = *(const v8h*)(bp + 8);
#pragma unroll
        for (int j = 0; j < 8; ++j) { b[j] = blo[j]; b[j+8] = bhi[j]; }
        acc = __builtin_amdgcn_wmma_f32_16x16x32_f16(false, a, false, b,
                                                     (short)0, acc, false, false);
    }

    // ---- epilogue: bias, store g, fused BN1 partial stats ----
    const float bias = fcb[col];
    const int rbase  = (lane >> 4) * 8;   // C/D layout: M = v + 8*(lane>>4)
    float s = 0.f, ss = 0.f;
#pragma unroll
    for (int v = 0; v < 8; ++v) {
        float gv = acc[v] + bias;
        size_t r = (size_t)tile*16 + rbase + v;
        g[r*(2*C_) + col] = gv;
        s  += gv;
        ss += gv*gv;
    }
    s  += __shfl_xor(s, 16, 32);
    ss += __shfl_xor(ss, 16, 32);
    if (lane < 16) {
        float* part = p1 + (size_t)(blockIdx.x & (NSLICE-1)) * (2*(2*C_));
        atomicAdd(&part[col], s);
        atomicAdd(&part[2*C_ + col], ss);
    }
}

// ---------------- finalize BN: scale/shift from sliced partials ----------------
__global__ void bn_finalize_kernel(const float* __restrict__ part, int nfeat,
                                   float cnt_inv, const float* __restrict__ gamma,
                                   const float* __restrict__ beta,
                                   float* __restrict__ scsh) {
    int f = blockIdx.x * blockDim.x + threadIdx.x;
    if (f >= nfeat) return;
    float s = 0.f, ss = 0.f;
    for (int i = 0; i < NSLICE; ++i) {
        s  += part[(size_t)i*2*nfeat + f];
        ss += part[(size_t)i*2*nfeat + nfeat + f];
    }
    float mean = s * cnt_inv;
    float var  = ss * cnt_inv - mean*mean;
    float sc   = gamma[f] * rsqrtf(var + EPSV);
    scsh[f]         = sc;
    scsh[nfeat + f] = beta[f] - mean * sc;
}

// ---------------- BN1 apply + gate + sum over M; fused BN2 stats ----------------
__global__ void conv_reduce_kernel(const float* __restrict__ g,
                                   const float* __restrict__ sc1,
                                   float* __restrict__ summed,
                                   float* __restrict__ p2) {
    int idx = blockIdx.x * 256 + threadIdx.x;             // N*C exactly
    int atom = idx >> 6, c = idx & 63;
    const float fs = sc1[c],        fb = sc1[2*C_ + c];
    const float cs = sc1[C_ + c],   cb = sc1[2*C_ + C_ + c];
    size_t base = (size_t)atom * M_ * (2*C_);
    float acc = 0.f;
#pragma unroll
    for (int m = 0; m < M_; ++m) {
        float gf = g[base + (size_t)m*(2*C_) + c]       * fs + fb;
        float gc = g[base + (size_t)m*(2*C_) + C_ + c]  * cs + cb;
        acc += sigmoidf_(gf) * softplusf(gc);
    }
    summed[idx] = acc;
    float* part = p2 + (size_t)(blockIdx.x & (NSLICE-1)) * (2*C_);
    atomicAdd(&part[c], acc);
    atomicAdd(&part[C_ + c], acc*acc);
}

// ---------------- BN2 apply + residual softplus ----------------
__global__ void conv_update_kernel(float* __restrict__ h, _Float16* __restrict__ h16,
                                   const float* __restrict__ summed,
                                   const float* __restrict__ sc2) {
    int idx = blockIdx.x * 256 + threadIdx.x;             // N*C exactly
    int c = idx & 63;
    float v = h[idx] + summed[idx]*sc2[c] + sc2[C_ + c];
    float r = softplusf(v);
    h[idx]   = r;
    h16[idx] = (_Float16)r;
}

// ---------------- per-crystal pooling (segment sum via atomics) ----------------
__global__ void pool_kernel(const float* __restrict__ h, const int* __restrict__ cidx,
                            float* __restrict__ crys, float* __restrict__ cnt) {
    int idx = blockIdx.x * 256 + threadIdx.x;             // N*C exactly
    int atom = idx >> 6, c = idx & 63;
    int cid = cidx[atom];
    atomicAdd(&crys[(size_t)cid*C_ + c], h[idx]);
    if (c == 0) atomicAdd(&cnt[cid], 1.f);
}

// ---------------- head: z = softplus(softplus(mean) @ head_W + head_b) ---------
__global__ void head_kernel(const float* __restrict__ crys, const float* __restrict__ cnt,
                            const float* __restrict__ hW, const float* __restrict__ hb,
                            float* __restrict__ z) {
    int idx = blockIdx.x * 256 + threadIdx.x;
    if (idx >= N0_ * H_) return;
    int n0 = idx >> 7, hh = idx & 127;
    float inv = 1.f / fmaxf(cnt[n0], 1.f);
    float acc = hb[hh];
#pragma unroll 4
    for (int c = 0; c < C_; ++c) {
        float cm = softplusf(crys[(size_t)n0*C_ + c] * inv);
        acc += cm * hW[(size_t)c*H_ + hh];
    }
    z[idx] = softplusf(acc);
}

// ---------------- output projection ----------------
__global__ void out_kernel(const float* __restrict__ z, const float* __restrict__ oW,
                           const float* __restrict__ ob, float* __restrict__ out) {
    int n0 = blockIdx.x * 256 + threadIdx.x;
    if (n0 >= N0_) return;
    float acc = ob[0];
#pragma unroll 8
    for (int h = 0; h < H_; ++h) acc += z[(size_t)n0*H_ + h] * oW[h];
    out[n0] = acc;
}

extern "C" void kernel_launch(void* const* d_in, const int* in_sizes, int n_in,
                              void* d_out, int out_size, void* d_ws, size_t ws_size,
                              hipStream_t stream) {
    const float* atom_fea = (const float*)d_in[0];
    const float* nbr_fea  = (const float*)d_in[1];
    const int*   nbr_idx  = (const int*)  d_in[2];
    const int*   cry_idx  = (const int*)  d_in[3];
    const float* emb_W    = (const float*)d_in[4];
    const float* emb_b    = (const float*)d_in[5];
    const float* fc_W     = (const float*)d_in[6];
    const float* fc_b     = (const float*)d_in[7];
    const float* bn1_g    = (const float*)d_in[8];
    const float* bn1_b    = (const float*)d_in[9];
    const float* bn2_g    = (const float*)d_in[10];
    const float* bn2_b    = (const float*)d_in[11];
    const float* head_W   = (const float*)d_in[12];
    const float* head_b   = (const float*)d_in[13];
    const float* out_W    = (const float*)d_in[14];
    const float* out_b    = (const float*)d_in[15];

    char* ws = (char*)d_ws;
    float*     g      = (float*)    (ws + OFF_G);
    float*     summed = (float*)    (ws + OFF_SUM);
    float*     h      = (float*)    (ws + OFF_H);
    _Float16*  h16    = (_Float16*) (ws + OFF_H16);
    _Float16*  wpack  = (_Float16*) (ws + OFF_WP);
    float*     p1     = (float*)    (ws + OFF_P1);
    float*     p2     = (float*)    (ws + OFF_P2);
    float*     sc1    = (float*)    (ws + OFF_SC1);
    float*     sc2    = (float*)    (ws + OFF_SC2);
    float*     crys   = (float*)    (ws + OFF_CRY);
    float*     cnt    = (float*)    (ws + OFF_CNT);
    float*     z      = (float*)    (ws + OFF_Z);

    // pack all three layers' fc weights into WMMA B-fragment layout
    packW_kernel<<<(NCONV_*KT_*CT_*32 + 255)/256, 256, 0, stream>>>(fc_W, wpack);
    // embedding
    embed_kernel<<<(N_*C_)/256, 256, 0, stream>>>(atom_fea, emb_W, emb_b, h, h16);

    for (int l = 0; l < NCONV_; ++l) {
        hipMemsetAsync(p1, 0, SZ_P1, stream);
        hipMemsetAsync(p2, 0, SZ_P2, stream);
        conv_fc_kernel<<<NTILE_, 256, 0, stream>>>(
            h16, nbr_idx, nbr_fea,
            wpack + (size_t)l * KT_*CT_*32*16,
            fc_b + (size_t)l * (2*C_), g, p1);
        bn_finalize_kernel<<<1, 128, 0, stream>>>(
            p1, 2*C_, 1.f/(float)NM_, bn1_g + (size_t)l*(2*C_), bn1_b + (size_t)l*(2*C_), sc1);
        conv_reduce_kernel<<<(N_*C_)/256, 256, 0, stream>>>(g, sc1, summed, p2);
        bn_finalize_kernel<<<1, 64, 0, stream>>>(
            p2, C_, 1.f/(float)N_, bn2_g + (size_t)l*C_, bn2_b + (size_t)l*C_, sc2);
        conv_update_kernel<<<(N_*C_)/256, 256, 0, stream>>>(h, h16, summed, sc2);
    }

    hipMemsetAsync(crys, 0, SZ_CRY, stream);
    hipMemsetAsync(cnt,  0, SZ_CNT, stream);
    pool_kernel<<<(N_*C_)/256, 256, 0, stream>>>(h, cry_idx, crys, cnt);
    head_kernel<<<(N0_*H_ + 255)/256, 256, 0, stream>>>(crys, cnt, head_W, head_b, z);
    out_kernel<<<(N0_ + 255)/256, 256, 0, stream>>>(z, out_W, out_b, (float*)d_out);
}
